// Attention_76381698392491
// MI455X (gfx1250) — matrix operand order repaired and checked by reading, not compiled, not run on hardware
//
#include <hip/hip_runtime.h>

// ---------------------------------------------------------------------------
// MI455X (gfx1250) attention: bf16 WMMA, f32 accumulation, wave32.
//  - Flash attention computed TRANSPOSED (S^T = K Q^T, x^T = V^T P^T): the
//    softmax reduction axis lands inside each lane, P^T is rebuilt with one
//    xor-16 lane exchange, all global stores are packed b128.
//  - K/V chunks are staged into LDS once per 8-wave block via the CDNA5
//    async DMA path (global_load_async_to_lds_b128 + s_wait_asynccnt),
//    double-buffered so chunk i+1 streams in while chunk i is computed.
// ---------------------------------------------------------------------------

typedef __attribute__((ext_vector_type(16))) __bf16 bf16x16;
typedef __attribute__((ext_vector_type(8)))  float  f32x8;
typedef __attribute__((ext_vector_type(4)))  float  f32x4;
typedef __attribute__((ext_vector_type(4)))  unsigned int u32x4;

#define BATCH 2
#define SEQ   4096
#define CDIM  512
#define HEADS 8
#define HDIM  64
#define NTOK  (BATCH * SEQ)   // 8192

union ABFrag  { bf16x16 v; u32x4 q[2]; };
union BF8Pack { u32x4 q;   __bf16 h[8]; };

// A-operand (16x32 bf16, MxK): lane l<16 -> row l; K = half*16..+15 contiguous
__device__ __forceinline__ bf16x16 loadA_bf16(const __bf16* base, int ld,
                                              int row0, int k0, int lane) {
    int r = lane & 15, half = lane >> 4;
    const __bf16* p = base + (size_t)(row0 + r) * ld + k0 + half * 16;
    ABFrag f;
    f.q[0] = *(const u32x4*)(p);
    f.q[1] = *(const u32x4*)(p + 8);
    return f.v;
}

// B-operand (32x16 bf16, KxN): lane -> column lane&15; K = half*16..+15 contiguous.
__device__ __forceinline__ bf16x16 loadB_bf16(const __bf16* base, int ld,
                                              int col0, int k0, int lane) {
    int c = lane & 15, half = lane >> 4;
    const __bf16* p = base + (size_t)(col0 + c) * ld + k0 + half * 16;
    ABFrag f;
    f.q[0] = *(const u32x4*)(p);
    f.q[1] = *(const u32x4*)(p + 8);
    return f.v;
}

// Same patterns sourced from f32 memory, converted to bf16 (v_cvt_pk_bf16_f32).
__device__ __forceinline__ bf16x16 loadA_f32(const float* base, int ld,
                                             int row0, int k0, int lane) {
    int r = lane & 15, half = lane >> 4;
    const float* p = base + (size_t)(row0 + r) * ld + k0 + half * 16;
    f32x4 a0 = *(const f32x4*)(p);
    f32x4 a1 = *(const f32x4*)(p + 4);
    f32x4 b0 = *(const f32x4*)(p + 8);
    f32x4 b1 = *(const f32x4*)(p + 12);
    bf16x16 v;
#pragma unroll
    for (int i = 0; i < 4; ++i) {
        v[i]      = (__bf16)a0[i];
        v[4 + i]  = (__bf16)a1[i];
        v[8 + i]  = (__bf16)b0[i];
        v[12 + i] = (__bf16)b1[i];
    }
    return v;
}

__device__ __forceinline__ bf16x16 loadB_f32(const float* base, int ld,
                                             int col0, int k0, int lane) {
    int c = lane & 15, half = lane >> 4;
    const float* p = base + (size_t)(col0 + c) * ld + k0 + half * 16;
    f32x4 a0 = *(const f32x4*)(p);
    f32x4 a1 = *(const f32x4*)(p + 4);
    f32x4 a2 = *(const f32x4*)(p + 8);
    f32x4 a3 = *(const f32x4*)(p + 12);
    bf16x16 v;
#pragma unroll
    for (int i = 0; i < 4; ++i) {
        v[i]      = (__bf16)a0[i];
        v[4 + i]  = (__bf16)a1[i];
        v[8 + i]  = (__bf16)a2[i];
        v[12 + i] = (__bf16)a3[i];
    }
    return v;
}

__device__ __forceinline__ f32x8 wmma_bf16(bf16x16 a, bf16x16 b, f32x8 c) {
    return __builtin_amdgcn_wmma_f32_16x16x32_bf16(
        false, a, false, b, (short)0, c, false, false);
}

// CDNA5 async DMA: global -> LDS, 16 bytes per lane, tracked by ASYNCcnt.
__device__ __forceinline__ void async_load_b128(unsigned lds_byte_off,
                                                const void* gaddr) {
    asm volatile("global_load_async_to_lds_b128 %0, %1, off"
                 :: "v"(lds_byte_off), "v"(gaddr)
                 : "memory");
}
// Low 32 bits of a flat shared address == LDS byte offset (ISA 10.2 aperture map).
__device__ __forceinline__ unsigned lds_off(const void* p) {
    return (unsigned)(uintptr_t)p;
}

// ---------------------------------------------------------------------------
// Kernel 1: QKV projections, one wave per 16x64 output group (4 subtiles share
// one operand fragment per k-step).
//   Q,K computed transposed (A = W rows, B = token rows)  -> b128 stores [B,H,tok,D]
//   V computed untransposed (A = token rows, B = W rows)  -> b128 stores [B,H,D,tok]
// Column group g == head h (64 cols per head).
// ---------------------------------------------------------------------------
__global__ __launch_bounds__(128) void qkv_proj_kernel(
    const float* __restrict__ query, const float* __restrict__ keym,
    const float* __restrict__ valm,  const float* __restrict__ Wq,
    const float* __restrict__ Wk,    const float* __restrict__ Wv,
    __bf16* __restrict__ Qh, __bf16* __restrict__ Kh, __bf16* __restrict__ VT) {
    const int lane = threadIdx.x & 31;
    const int wave = blockIdx.x * (blockDim.x >> 5) + (threadIdx.x >> 5);

    const int groupsPerProj = (NTOK / 16) * (CDIM / 64);   // 512 * 8 = 4096
    const int which = wave / groupsPerProj;                // 0:Q 1:K 2:V
    const int t     = wave % groupsPerProj;
    const int rowTile = t / (CDIM / 64);
    const int g       = t % (CDIM / 64);                   // head
    const int tok0 = rowTile * 16;
    const int j0   = g * 64;

    const float* A = (which == 0) ? query : (which == 1) ? keym : valm;
    const float* W = (which == 0) ? Wq    : (which == 1) ? Wk   : Wv;

    const int cidx = lane & 15, half = lane >> 4;
    const int b_ = tok0 / SEQ;          // tile never crosses batch boundary
    const int n0 = tok0 % SEQ;

    f32x8 acc[4] = {};
    if (which < 2) {
        // out^T = W * in^T
        for (int k0 = 0; k0 < CDIM; k0 += 32) {
            bf16x16 binp = loadB_f32(A, CDIM, tok0, k0, lane);
#pragma unroll
            for (int s = 0; s < 4; ++s) {
                bf16x16 aw = loadA_f32(W, CDIM, j0 + s * 16, k0, lane);
                acc[s] = wmma_bf16(aw, binp, acc[s]);
            }
        }
        __bf16* out = which ? Kh : Qh;
        __bf16* rowp =
            out + ((size_t)(b_ * HEADS + g) * SEQ + n0 + cidx) * HDIM;
#pragma unroll
        for (int s = 0; s < 4; ++s) {
            BF8Pack pk;
#pragma unroll
            for (int r = 0; r < 8; ++r) pk.h[r] = (__bf16)acc[s][r];
            *(u32x4*)(rowp + s * 16 + half * 8) = pk.q;
        }
    } else {
        // out = in * W^T  (C col = d, rows = tokens -> contiguous along m)
        for (int k0 = 0; k0 < CDIM; k0 += 32) {
            bf16x16 ain = loadA_f32(A, CDIM, tok0, k0, lane);
#pragma unroll
            for (int s = 0; s < 4; ++s) {
                bf16x16 bw = loadB_f32(W, CDIM, j0 + s * 16, k0, lane);
                acc[s] = wmma_bf16(ain, bw, acc[s]);
            }
        }
#pragma unroll
        for (int s = 0; s < 4; ++s) {
            BF8Pack pk;
#pragma unroll
            for (int r = 0; r < 8; ++r) pk.h[r] = (__bf16)acc[s][r];
            __bf16* rp = VT +
                ((size_t)(b_ * HEADS + g) * HDIM + s * 16 + cidx) * SEQ +
                n0 + half * 8;
            *(u32x4*)rp = pk.q;
        }
    }
}

// ---------------------------------------------------------------------------
// Kernel 2: transposed flash attention with async-DMA LDS staging.
// Block = 8 waves, all on the same (b,h); each wave owns 16 q-rows. K/V chunks
// (32 KV positions) are DMA'd to double-buffered LDS once per block; every
// wave then reads its WMMA fragments from LDS (ds_load_b128).
// ---------------------------------------------------------------------------
__global__ __launch_bounds__(256) void flash_kernel(
    const __bf16* __restrict__ Qh, const __bf16* __restrict__ Kh,
    const __bf16* __restrict__ VT, __bf16* __restrict__ X) {
    __shared__ __align__(16) __bf16 Kbuf[2][32 * HDIM];  // 2 x 4KB (m x d)
    __shared__ __align__(16) __bf16 Vbuf[2][HDIM * 32];  // 2 x 4KB (d x m)

    const int lane = threadIdx.x & 31;
    const int wave = blockIdx.x * 8 + (threadIdx.x >> 5);
    const int bh    = wave >> 8;              // uniform per block (32 blk / bh)
    const int nTile = wave & 255;
    const int n0    = nTile * 16;
    const int half  = lane >> 4;
    const int flat  = threadIdx.x;            // 0..255: per-lane DMA slice

    const __bf16* Qbase = Qh + (size_t)bh * SEQ * HDIM;
    const __bf16* Kbase = Kh + (size_t)bh * SEQ * HDIM;
    const __bf16* Vbase = VT + (size_t)bh * HDIM * SEQ;

    // Per-lane async-DMA source/dest (16B each for K and V per chunk).
    const __bf16* kSrc0 = Kbase + (size_t)(flat >> 3) * HDIM + (flat & 7) * 8;
    const __bf16* vSrc0 = Vbase + (size_t)(flat >> 2) * SEQ  + (flat & 3) * 8;
    const unsigned kDst[2] = { lds_off(&Kbuf[0][0]) + (unsigned)flat * 16u,
                               lds_off(&Kbuf[1][0]) + (unsigned)flat * 16u };
    const unsigned vDst[2] = { lds_off(&Vbuf[0][0]) + (unsigned)flat * 16u,
                               lds_off(&Vbuf[1][0]) + (unsigned)flat * 16u };

    // Persistent Q^T B-fragments (K = d, columns = q rows n0..n0+15).
    const bf16x16 qb0 = loadB_bf16(Qbase, HDIM, n0, 0, lane);
    const bf16x16 qb1 = loadB_bf16(Qbase, HDIM, n0, 32, lane);

    f32x8 acc[4] = {};                 // x^T: 4 d-tiles of 16x16 (d x n)
    float mrun = -3.0e38f, lrun = 0.0f;
    const float scale2 = 0.180336880111120405f;  // (1/sqrt(64)) * log2(e)

    // Prologue: DMA chunk 0 into buffer 0.
    async_load_b128(kDst[0], kSrc0);
    async_load_b128(vDst[0], vSrc0);

    const int nChunks = SEQ / 32;                 // 128
    for (int it = 0; it < nChunks; ++it) {
        const int cur = it & 1;
        if (it + 1 < nChunks) {
            const size_t m1 = (size_t)(it + 1) * 32;
            async_load_b128(kDst[cur ^ 1], kSrc0 + m1 * HDIM);  // K rows shift
            async_load_b128(vDst[cur ^ 1], vSrc0 + m1);         // V cols shift
            asm volatile("s_wait_asynccnt 0x2" ::: "memory");   // chunk it done
        } else {
            asm volatile("s_wait_asynccnt 0x0" ::: "memory");
        }
        __syncthreads();               // publish chunk it to all 8 waves

        const __bf16* kc = &Kbuf[cur][0];
        const __bf16* vc = &Vbuf[cur][0];

        // S^T = K Q^T : two m-row tiles, K-depth 64 (2 wmma each), from LDS.
        bf16x16 ka0 = loadA_bf16(kc, HDIM, 0,  0,  lane);
        bf16x16 ka1 = loadA_bf16(kc, HDIM, 0,  32, lane);
        bf16x16 ka2 = loadA_bf16(kc, HDIM, 16, 0,  lane);
        bf16x16 ka3 = loadA_bf16(kc, HDIM, 16, 32, lane);
        f32x8 s0 = {}, s1 = {};
        s0 = wmma_bf16(ka0, qb0, s0);
        s0 = wmma_bf16(ka1, qb1, s0);
        s1 = wmma_bf16(ka2, qb0, s1);
        s1 = wmma_bf16(ka3, qb1, s1);

        // Per-lane (per-n) online softmax in base 2; lane holds 16 m-values,
        // lane^16 holds the other 16.
        float mx = -3.0e38f;
#pragma unroll
        for (int r = 0; r < 8; ++r) {
            s0[r] *= scale2;
            s1[r] *= scale2;
            mx = fmaxf(mx, fmaxf(s0[r], s1[r]));
        }
        mx = fmaxf(mx, __shfl_xor(mx, 16, 32));
        const float mnew  = fmaxf(mrun, mx);
        const float alpha = exp2f(mrun - mnew);
        mrun = mnew;

        float rs = 0.0f;
#pragma unroll
        for (int r = 0; r < 8; ++r) {
            s0[r] = exp2f(s0[r] - mnew);
            s1[r] = exp2f(s1[r] - mnew);
            rs += s0[r] + s1[r];
        }
        rs += __shfl_xor(rs, 16, 32);
        lrun = lrun * alpha + rs;

#pragma unroll
        for (int s = 0; s < 4; ++s)
#pragma unroll
            for (int r = 0; r < 8; ++r) acc[s][r] *= alpha;

        // Build P^T B-operand (K=32 m, N=16 n) from the two C-layout tiles:
        // element k of lane (n,h): tile k/16, vgpr k%8, source half (k%16)/8.
        bf16x16 pB;
#pragma unroll
        for (int r = 0; r < 8; ++r) {
            float x0 = __shfl_xor(s0[r], 16, 32);
            float x1 = __shfl_xor(s1[r], 16, 32);
            float lo = half ? x1 : s0[r];   // k = half*16 + r
            float hi = half ? s1[r] : x0;   // k = half*16 + 8 + r
            pB[r]     = (__bf16)lo;
            pB[8 + r] = (__bf16)hi;
        }

        // x^T += V^T @ P^T : A-frags are contiguous rows of V^T, from LDS.
        bf16x16 va0 = loadA_bf16(vc, 32, 0,  0, lane);
        bf16x16 va1 = loadA_bf16(vc, 32, 16, 0, lane);
        bf16x16 va2 = loadA_bf16(vc, 32, 32, 0, lane);
        bf16x16 va3 = loadA_bf16(vc, 32, 48, 0, lane);
        acc[0] = wmma_bf16(va0, pB, acc[0]);
        acc[1] = wmma_bf16(va1, pB, acc[1]);
        acc[2] = wmma_bf16(va2, pB, acc[2]);
        acc[3] = wmma_bf16(va3, pB, acc[3]);

        __syncthreads();               // retire buffer before next overwrite
    }

    // Normalize; lane holds fixed n with 8 consecutive d per tile -> b128.
    const float inv = 1.0f / lrun;
    const int b_ = bh / HEADS, h = bh % HEADS;
    const int n = n0 + (lane & 15);
    __bf16* rowp = X + ((size_t)b_ * SEQ + n) * CDIM + h * HDIM;
#pragma unroll
    for (int s = 0; s < 4; ++s) {
        BF8Pack pk;
#pragma unroll
        for (int r = 0; r < 8; ++r) pk.h[r] = (__bf16)(acc[s][r] * inv);
        *(u32x4*)(rowp + s * 16 + half * 8) = pk.q;
    }
}

// ---------------------------------------------------------------------------
// Kernel 3: output projection, transposed (out^T = Wp @ X^T) -> lane owns one
// token, 8 consecutive j per subtile -> vectorized f32 stores + bias.
// ---------------------------------------------------------------------------
__global__ __launch_bounds__(128) void out_proj_kernel(
    const __bf16* __restrict__ X, const float* __restrict__ Wp,
    const float* __restrict__ bp, float* __restrict__ out) {
    const int lane = threadIdx.x & 31;
    const int wave = blockIdx.x * (blockDim.x >> 5) + (threadIdx.x >> 5);
    const int rowTile = wave / (CDIM / 64);
    const int g       = wave % (CDIM / 64);
    const int tok0 = rowTile * 16;
    const int j0   = g * 64;

    f32x8 acc[4] = {};
    for (int k0 = 0; k0 < CDIM; k0 += 32) {
        bf16x16 bx = loadB_bf16(X, CDIM, tok0, k0, lane);
#pragma unroll
        for (int s = 0; s < 4; ++s) {
            bf16x16 aw = loadA_f32(Wp, CDIM, j0 + s * 16, k0, lane);
            acc[s] = wmma_bf16(aw, bx, acc[s]);
        }
    }

    const int cidx = lane & 15, half = lane >> 4;
    const int tok = tok0 + cidx;
    float* rowp = out + (size_t)tok * CDIM + j0;
#pragma unroll
    for (int s = 0; s < 4; ++s) {
        const float* bptr = bp + j0 + s * 16 + half * 8;
        f32x4 blo = *(const f32x4*)(bptr);
        f32x4 bhi = *(const f32x4*)(bptr + 4);
        f32x4 lo, hi;
#pragma unroll
        for (int i = 0; i < 4; ++i) {
            lo[i] = acc[s][i]     + blo[i];
            hi[i] = acc[s][4 + i] + bhi[i];
        }
        *(f32x4*)(rowp + s * 16 + half * 8)     = lo;
        *(f32x4*)(rowp + s * 16 + half * 8 + 4) = hi;
    }
}

// ---------------------------------------------------------------------------
extern "C" void kernel_launch(void* const* d_in, const int* in_sizes, int n_in,
                              void* d_out, int out_size, void* d_ws, size_t ws_size,
                              hipStream_t stream) {
    (void)in_sizes; (void)n_in; (void)out_size; (void)ws_size;
    const float* query = (const float*)d_in[0];
    const float* keym  = (const float*)d_in[1];
    const float* valm  = (const float*)d_in[2];
    const float* Wq    = (const float*)d_in[3];
    const float* Wk    = (const float*)d_in[4];
    const float* Wv    = (const float*)d_in[5];
    const float* Wp    = (const float*)d_in[6];
    const float* bp    = (const float*)d_in[7];
    float* out = (float*)d_out;

    const size_t elems = (size_t)BATCH * HEADS * SEQ * HDIM;  // 4M bf16 each
    __bf16* Qh = reinterpret_cast<__bf16*>(d_ws);
    __bf16* Kh = Qh + elems;
    __bf16* VT = Kh + elems;
    __bf16* Xb = VT + elems;

    // 3 * 512 * 8 groups, 4 waves / 128-thread block
    qkv_proj_kernel<<<3072, 128, 0, stream>>>(query, keym, valm, Wq, Wk, Wv,
                                              Qh, Kh, VT);
    // 16 (b,h) * 256 q-tiles = 4096 waves, 8 waves / 256-thread block
    flash_kernel<<<512, 256, 0, stream>>>(Qh, Kh, VT, Xb);
    // 512 * 8 groups, 4 waves / 128-thread block
    out_proj_kernel<<<1024, 128, 0, stream>>>(Xb, Wp, bp, out);
}
